// poolMGN_38345468018705
// MI455X (gfx1250) — matrix operand order repaired and checked by reading, not compiled, not run on hardware
//
#include <hip/hip_runtime.h>

typedef __bf16 bf16_t;
typedef bf16_t v16bf __attribute__((ext_vector_type(16)));
typedef float  v8f   __attribute__((ext_vector_type(8)));

#define N_NODES 10000
#define N_EDGES 160000
#define HID 128
#define WAVES 2
#define BLOCK (WAVES * 32)

__device__ __forceinline__ float  bf2f(bf16_t x) { return (float)x; }
__device__ __forceinline__ bf16_t f2bf(float x)  { return (bf16_t)x; }

// CDNA5 async copy: global -> LDS with no VGPR data path (ASYNCcnt tracked).
__device__ __forceinline__ void async_g2l_b128(const void* gptr, void* lptr) {
  const unsigned long long ga = (unsigned long long)(size_t)gptr;
  const unsigned la = (unsigned)(size_t)lptr;   // low 32 bits = LDS byte offset
  asm volatile("global_load_async_to_lds_b128 %0, %1, off"
               :: "v"(la), "v"(ga));
}
__device__ __forceinline__ void wait_async0() {
  asm volatile("s_wait_asynccnt 0x0" ::: "memory");
}

union Frag { v16bf v; uint4 q[2]; };

struct MlpArgs {
  const float*  a0;   // fp32 attr source (node_attr / edge_attr)
  const bf16_t* x;    // bf16 node state
  const float*  a2;   // fp32 aux source (g_bcast or agg)
  const bf16_t* e;    // bf16 edge state
  const int*    src;
  const int*    dst;
  const bf16_t* W1; const float* b1;
  const bf16_t* W2; const float* b2;
  const float*  lng; const float* lnb;
  bf16_t* out_bf;
  float*  out_f;
  float*  agg;
  int M;
};

// MODE 0: edge_enc   (K=8 ->32 pad,  LN, out e bf16)
// MODE 1: global_enc (K=16->32 pad,  no LN, out g f32)
// MODE 2: node_enc   (K=144->160,    LN, out x bf16)
// MODE 3: edge layer (K=384,         LN, residual e, out e bf16 + agg scatter)
// MODE 4: node layer (K=256,         LN, residual x, out x bf16)
template <int MODE>
__global__ __launch_bounds__(BLOCK) void mlp2_kernel(MlpArgs a) {
  constexpr int  KPAD = (MODE <= 1) ? 32 : (MODE == 2 ? 160 : (MODE == 3 ? 384 : 256));
  constexpr int  KT   = KPAD / 32;
  constexpr bool LN   = (MODE != 1);
  constexpr int  RES  = (MODE == 3) ? 256 : (MODE == 4 ? 0 : -1);
  constexpr bool OUTF = (MODE == 1);
  constexpr bool SCAT = (MODE == 3);

  __shared__ __align__(16) bf16_t A_lds[WAVES * 16 * KPAD];
  __shared__ __align__(16) bf16_t H_lds[WAVES * 16 * HID];

  const int lane = threadIdx.x & 31;
  const int wave = threadIdx.x >> 5;
  const int tile = blockIdx.x * WAVES + wave;
  if (tile * 16 >= a.M) return;           // wave-uniform: EXEC stays all-1s
  const int row0 = tile * 16;

  bf16_t* Arow = &A_lds[wave * 16 * KPAD];
  bf16_t* Hrow = &H_lds[wave * 16 * HID];

  __builtin_prefetch(a.W1, 0, 3);
  __builtin_prefetch(a.W2, 0, 3);

  // ---------------- stage A rows into LDS (per-wave, no block barrier) -----
  if constexpr (MODE == 3) {
    // row = x[src] | x[dst] | e : 48 b128 chunks per row, async global->LDS
    for (int idx = lane; idx < 16 * 48; idx += 32) {   // 24 iters, uniform
      const int r = idx / 48, c = idx % 48;
      const int seg = c >> 4, off = c & 15;
      const int er = row0 + r;
      const bf16_t* gp;
      if (seg == 0)      gp = a.x + (size_t)a.src[er] * HID;
      else if (seg == 1) gp = a.x + (size_t)a.dst[er] * HID;
      else               gp = a.e + (size_t)er * HID;
      async_g2l_b128(gp + off * 8, Arow + r * KPAD + c * 8);
    }
    wait_async0();
  } else if constexpr (MODE == 4) {
    // row = x | bf16(agg): x part async, agg part converted via VALU
    for (int idx = lane; idx < 16 * 16; idx += 32) {   // 8 iters, uniform
      const int r = idx >> 4, off = idx & 15;
      async_g2l_b128(a.x + (size_t)(row0 + r) * HID + off * 8,
                     Arow + r * KPAD + off * 8);
    }
    for (int idx = lane; idx < 16 * HID; idx += 32) {
      const int r = idx >> 7, c = idx & 127;
      Arow[r * KPAD + HID + c] = f2bf(a.a2[(size_t)(row0 + r) * HID + c]);
    }
    wait_async0();
  } else if constexpr (MODE == 2) {
    // row = node_attr(16) | g_bcast(128) | zeropad(16)
    for (int idx = lane; idx < 16 * KPAD; idx += 32) {
      const int r = idx / KPAD, c = idx % KPAD;
      float v = 0.f;
      if (c < 16)       v = a.a0[(size_t)(row0 + r) * 16 + c];
      else if (c < 144) v = a.a2[(size_t)(row0 + r) * HID + (c - 16)];
      Arow[r * KPAD + c] = f2bf(v);
    }
  } else {
    constexpr int DIN = (MODE == 0) ? 8 : 16;
    for (int idx = lane; idx < 16 * KPAD; idx += 32) {
      const int r = idx >> 5, c = idx & 31;
      const float v = (c < DIN) ? a.a0[(size_t)(row0 + r) * DIN + c] : 0.f;
      Arow[r * KPAD + c] = f2bf(v);
    }
  }

  const int lanelo = lane & 15, half = lane >> 4;

  // ---------------- GEMM1: [16,KPAD] x [KPAD,128] -> relu -> H -------------
  v8f acc[8] = {};
  for (int kt = 0; kt < KT; ++kt) {
    Frag af;
    const bf16_t* ab = Arow + lanelo * KPAD + kt * 32 + half * 8;
    af.q[0] = *(const uint4*)(ab);
    af.q[1] = *(const uint4*)(ab + 16);
#pragma unroll
    for (int nt = 0; nt < 8; ++nt) {
      Frag bfr;
      const bf16_t* bp = a.W1 + (size_t)(kt * 32 + half * 8) * HID + nt * 16 + lanelo;
      for (int s = 0; s < 16; ++s) bfr.v[s] = bp[(size_t)((s & 7) + (s >> 3) * 16) * HID];
      acc[nt] = __builtin_amdgcn_wmma_f32_16x16x32_bf16(
          false, af.v, false, bfr.v, (short)0, acc[nt], false, false);
    }
  }
#pragma unroll
  for (int nt = 0; nt < 8; ++nt) {
    const float bias = a.b1[nt * 16 + lanelo];
#pragma unroll
    for (int r = 0; r < 8; ++r) {
      const float v = fmaxf(acc[nt][r] + bias, 0.f);
      Hrow[(r + 8 * half) * HID + nt * 16 + lanelo] = f2bf(v);
    }
  }

  // ---------------- GEMM2: [16,128] x [128,128] ----------------------------
  v8f acc2[8] = {};
  for (int kt = 0; kt < 4; ++kt) {
    Frag af;
    const bf16_t* ab = Hrow + lanelo * HID + kt * 32 + half * 8;
    af.q[0] = *(const uint4*)(ab);
    af.q[1] = *(const uint4*)(ab + 16);
#pragma unroll
    for (int nt = 0; nt < 8; ++nt) {
      Frag bfr;
      const bf16_t* bp = a.W2 + (size_t)(kt * 32 + half * 8) * HID + nt * 16 + lanelo;
      for (int s = 0; s < 16; ++s) bfr.v[s] = bp[(size_t)((s & 7) + (s >> 3) * 16) * HID];
      acc2[nt] = __builtin_amdgcn_wmma_f32_16x16x32_bf16(
          false, af.v, false, bfr.v, (short)0, acc2[nt], false, false);
    }
  }

  // ---------------- epilogue: bias -> LN -> residual -> store / scatter ----
#pragma unroll
  for (int r = 0; r < 8; ++r) {
    const int lrow = r + 8 * half;
    const int grow = row0 + lrow;
    float v[8];
    float s = 0.f, ss = 0.f;
#pragma unroll
    for (int nt = 0; nt < 8; ++nt) {
      v[nt] = acc2[nt][r] + a.b2[nt * 16 + lanelo];
      s += v[nt];
      ss += v[nt] * v[nt];
    }
    if constexpr (LN) {
#pragma unroll
      for (int off = 8; off >= 1; off >>= 1) {
        s  += __shfl_xor(s,  off, 16);
        ss += __shfl_xor(ss, off, 16);
      }
      const float mean = s * (1.f / 128.f);
      const float var  = ss * (1.f / 128.f) - mean * mean;
      const float rstd = rsqrtf(var + 1e-5f);
#pragma unroll
      for (int nt = 0; nt < 8; ++nt) {
        const int col = nt * 16 + lanelo;
        v[nt] = (v[nt] - mean) * rstd * a.lng[col] + a.lnb[col];
      }
    }
    if constexpr (RES >= 0) {
#pragma unroll
      for (int nt = 0; nt < 8; ++nt)
        v[nt] += bf2f(Arow[lrow * KPAD + RES + nt * 16 + lanelo]);
    }
    if constexpr (OUTF) {
#pragma unroll
      for (int nt = 0; nt < 8; ++nt)
        a.out_f[(size_t)grow * HID + nt * 16 + lanelo] = v[nt];
    } else {
#pragma unroll
      for (int nt = 0; nt < 8; ++nt)
        a.out_bf[(size_t)grow * HID + nt * 16 + lanelo] = f2bf(v[nt]);
    }
    if constexpr (SCAT) {
      const int d = a.dst[grow];
#pragma unroll
      for (int nt = 0; nt < 8; ++nt)
        atomicAdd(&a.agg[(size_t)d * HID + nt * 16 + lanelo], v[nt]);
    }
  }
}

// --------------------------- helper kernels --------------------------------
__global__ void convert_pad_kernel(const float* __restrict__ s,
                                   bf16_t* __restrict__ d, int Ksrc, int Kdst) {
  const int i = blockIdx.x * blockDim.x + threadIdx.x;
  if (i < Kdst * HID) d[i] = f2bf(((i >> 7) < Ksrc) ? s[i] : 0.f);
}

__global__ void fill_f32_kernel(float* p, int n, float v) {
  const int i = blockIdx.x * blockDim.x + threadIdx.x;
  if (i < n) p[i] = v;
}

__global__ void pool_sum_kernel(const float* __restrict__ g,
                                const int* __restrict__ batch,
                                float* sums, float* cnt, int N) {
  const int i = blockIdx.x * blockDim.x + threadIdx.x;
  if (i < N * HID) {
    const int nd = i >> 7;
    atomicAdd(&sums[batch[nd] * HID + (i & 127)], g[i]);
  }
  if (i < N) atomicAdd(&cnt[batch[i]], 1.0f);
}

__global__ void bcast_kernel(const float* __restrict__ sums,
                             const float* __restrict__ cnt,
                             const int* __restrict__ batch, float* gb, int N) {
  const int i = blockIdx.x * blockDim.x + threadIdx.x;
  if (i < N * HID) {
    const int b = batch[i >> 7];
    gb[i] = sums[b * HID + (i & 127)] / fmaxf(cnt[b], 1.f);
  }
}

__global__ void dec_hidden_kernel(const bf16_t* __restrict__ x,
                                  const float* __restrict__ W,
                                  const float* __restrict__ b, float* h, int N) {
  const int i = blockIdx.x * blockDim.x + threadIdx.x;
  if (i >= N * HID) return;
  const int nd = i >> 7, c = i & 127;
  float s = b[c];
  for (int k = 0; k < HID; ++k) s += bf2f(x[(size_t)nd * HID + k]) * W[k * HID + c];
  h[i] = fmaxf(s, 0.f);
}

__global__ void dec_out_kernel(const float* __restrict__ h,
                               const float* __restrict__ W,
                               const float* __restrict__ b, float* out, int N) {
  const int i = blockIdx.x * blockDim.x + threadIdx.x;
  if (i >= N * 4) return;
  const int nd = i >> 2, o = i & 3;
  float s = b[o];
  for (int k = 0; k < HID; ++k) s += h[(size_t)nd * HID + k] * W[k * 4 + o];
  out[i] = s;
}

// --------------------------- host orchestration ----------------------------
extern "C" void kernel_launch(void* const* d_in, const int* in_sizes, int n_in,
                              void* d_out, int out_size, void* d_ws, size_t ws_size,
                              hipStream_t stream) {
  (void)in_sizes; (void)n_in; (void)out_size; (void)ws_size;
  const int N = N_NODES, E = N_EDGES;

  const float* node_attr = (const float*)d_in[0];
  const float* edge_attr = (const float*)d_in[1];
  const int*   ei        = (const int*)d_in[2];
  const int*   srcI      = ei;
  const int*   dstI      = ei + E;
  const int*   batch     = (const int*)d_in[3];

  // params flattened: dicts alphabetical -> decoder, edge_enc, global_enc, layers, node_enc
  const float* dec_hW = (const float*)d_in[4];
  const float* dec_hb = (const float*)d_in[5];
  const float* dec_oW = (const float*)d_in[6];
  const float* dec_ob = (const float*)d_in[7];
  const float* ee_hW  = (const float*)d_in[8];
  const float* ee_hb  = (const float*)d_in[9];
  const float* ee_g   = (const float*)d_in[10];
  const float* ee_b   = (const float*)d_in[11];
  const float* ee_oW  = (const float*)d_in[12];
  const float* ee_ob  = (const float*)d_in[13];
  const float* ge_hW  = (const float*)d_in[14];
  const float* ge_hb  = (const float*)d_in[15];
  const float* ge_oW  = (const float*)d_in[16];
  const float* ge_ob  = (const float*)d_in[17];
  const int LBASE = 18;                 // 15 layers x 12 leaves
  const float* ne_hW  = (const float*)d_in[198];
  const float* ne_hb  = (const float*)d_in[199];
  const float* ne_g   = (const float*)d_in[200];
  const float* ne_b   = (const float*)d_in[201];
  const float* ne_oW  = (const float*)d_in[202];
  const float* ne_ob  = (const float*)d_in[203];

  char* ws = (char*)d_ws;
  size_t off = 0;
  auto alloc = [&](size_t bytes) -> char* {
    char* p = ws + off;
    off += (bytes + 255) & ~(size_t)255;
    return p;
  };
  bf16_t* x    = (bf16_t*)alloc((size_t)N * HID * 2);
  bf16_t* e    = (bf16_t*)alloc((size_t)E * HID * 2);
  float*  agg  = (float*)alloc((size_t)N * HID * 4);
  float*  g    = (float*)alloc((size_t)N * HID * 4);
  float*  gb   = (float*)alloc((size_t)N * HID * 4);
  float*  hd   = (float*)alloc((size_t)N * HID * 4);
  float*  sums = (float*)alloc(4 * HID * 4);
  float*  cnt  = (float*)alloc(64);
  bf16_t* wge1 = (bf16_t*)alloc(32 * HID * 2);
  bf16_t* wge2 = (bf16_t*)alloc(HID * HID * 2);
  bf16_t* wee1 = (bf16_t*)alloc(32 * HID * 2);
  bf16_t* wee2 = (bf16_t*)alloc(HID * HID * 2);
  bf16_t* wne1 = (bf16_t*)alloc(160 * HID * 2);
  bf16_t* wne2 = (bf16_t*)alloc(HID * HID * 2);
  bf16_t* wl[15][4];
  for (int i = 0; i < 15; ++i) {
    wl[i][0] = (bf16_t*)alloc(384 * HID * 2);
    wl[i][1] = (bf16_t*)alloc(HID * HID * 2);
    wl[i][2] = (bf16_t*)alloc(256 * HID * 2);
    wl[i][3] = (bf16_t*)alloc(HID * HID * 2);
  }

  auto cvt = [&](const float* s, bf16_t* d, int Ks, int Kd) {
    const int n = Kd * HID;
    convert_pad_kernel<<<(n + 255) / 256, 256, 0, stream>>>(s, d, Ks, Kd);
  };
  cvt(ge_hW, wge1, 16, 32);    cvt(ge_oW, wge2, 128, 128);
  cvt(ee_hW, wee1, 8, 32);     cvt(ee_oW, wee2, 128, 128);
  cvt(ne_hW, wne1, 144, 160);  cvt(ne_oW, wne2, 128, 128);
  for (int i = 0; i < 15; ++i) {
    cvt((const float*)d_in[LBASE + 12 * i + 0], wl[i][0], 384, 384);
    cvt((const float*)d_in[LBASE + 12 * i + 4], wl[i][1], 128, 128);
    cvt((const float*)d_in[LBASE + 12 * i + 6], wl[i][2], 256, 256);
    cvt((const float*)d_in[LBASE + 12 * i + 10], wl[i][3], 128, 128);
  }

  const int nblk = (N / 16 + WAVES - 1) / WAVES;   // 313
  const int eblk = (E / 16 + WAVES - 1) / WAVES;   // 5000

  // 1. global encoder -> g (fp32)
  {
    MlpArgs ar{};
    ar.a0 = node_attr; ar.W1 = wge1; ar.b1 = ge_hb; ar.W2 = wge2; ar.b2 = ge_ob;
    ar.out_f = g; ar.M = N;
    mlp2_kernel<1><<<nblk, BLOCK, 0, stream>>>(ar);
  }
  // 2. per-graph mean pool + broadcast
  fill_f32_kernel<<<2, 256, 0, stream>>>(sums, 4 * HID, 0.f);
  fill_f32_kernel<<<1, 64, 0, stream>>>(cnt, 4, 0.f);
  pool_sum_kernel<<<(N * HID + 255) / 256, 256, 0, stream>>>(g, batch, sums, cnt, N);
  bcast_kernel<<<(N * HID + 255) / 256, 256, 0, stream>>>(sums, cnt, batch, gb, N);
  // 3. node encoder -> x
  {
    MlpArgs ar{};
    ar.a0 = node_attr; ar.a2 = gb;
    ar.W1 = wne1; ar.b1 = ne_hb; ar.W2 = wne2; ar.b2 = ne_ob;
    ar.lng = ne_g; ar.lnb = ne_b; ar.out_bf = x; ar.M = N;
    mlp2_kernel<2><<<nblk, BLOCK, 0, stream>>>(ar);
  }
  // 4. edge encoder -> e
  {
    MlpArgs ar{};
    ar.a0 = edge_attr;
    ar.W1 = wee1; ar.b1 = ee_hb; ar.W2 = wee2; ar.b2 = ee_ob;
    ar.lng = ee_g; ar.lnb = ee_b; ar.out_bf = e; ar.M = E;
    mlp2_kernel<0><<<eblk, BLOCK, 0, stream>>>(ar);
  }
  // 5. message-passing layers
  for (int i = 0; i < 15; ++i) {
    const int lb = LBASE + 12 * i;
    fill_f32_kernel<<<(N * HID + 255) / 256, 256, 0, stream>>>(agg, N * HID, 0.f);
    {
      MlpArgs ar{};
      ar.x = x; ar.e = e; ar.src = srcI; ar.dst = dstI;
      ar.W1 = wl[i][0]; ar.b1 = (const float*)d_in[lb + 1];
      ar.lng = (const float*)d_in[lb + 2]; ar.lnb = (const float*)d_in[lb + 3];
      ar.W2 = wl[i][1]; ar.b2 = (const float*)d_in[lb + 5];
      ar.out_bf = e; ar.agg = agg; ar.M = E;
      mlp2_kernel<3><<<eblk, BLOCK, 0, stream>>>(ar);
    }
    {
      MlpArgs ar{};
      ar.x = x; ar.a2 = agg;
      ar.W1 = wl[i][2]; ar.b1 = (const float*)d_in[lb + 7];
      ar.lng = (const float*)d_in[lb + 8]; ar.lnb = (const float*)d_in[lb + 9];
      ar.W2 = wl[i][3]; ar.b2 = (const float*)d_in[lb + 11];
      ar.out_bf = x; ar.M = N;
      mlp2_kernel<4><<<nblk, BLOCK, 0, stream>>>(ar);
    }
  }
  // 6. decoder
  dec_hidden_kernel<<<(N * HID + 255) / 256, 256, 0, stream>>>(x, dec_hW, dec_hb, hd, N);
  dec_out_kernel<<<(N * 4 + 255) / 256, 256, 0, stream>>>(hd, dec_oW, dec_ob, (float*)d_out, N);
}
